// Lstm_72885595013405
// MI455X (gfx1250) — compile-verified, matrix operand order checked by reading, not run on hardware
//
#include <hip/hip_runtime.h>
#include <hip/hip_bf16.h>
#include <stdint.h>

#define B_TOT 512
#define T_TOT 1024
#define I_DIM 64
#define H_DIM 128
#define G_DIM 512            // 4*H
#define K_DIM 192            // I + H
#define XH_STRIDE 200        // padded LDS row stride (bf16 elems)

typedef __attribute__((ext_vector_type(16))) __bf16        v16bf;
typedef __attribute__((ext_vector_type(8)))  float         v8f;
typedef __attribute__((ext_vector_type(8)))  unsigned int  v8u;
typedef __attribute__((ext_vector_type(4)))  int           v4i;
typedef __attribute__((ext_vector_type(4)))  __bf16        v4bf;

// Address-space qualified element types for the async-LDS builtin.
typedef __attribute__((address_space(1))) v4i v4i_as1;
typedef __attribute__((address_space(3))) v4i v4i_as3;

#if defined(__has_builtin)
#if __has_builtin(__builtin_amdgcn_global_load_async_to_lds_b128) && \
    __has_builtin(__builtin_amdgcn_s_wait_asynccnt)
#define USE_ASYNC_LDS 1
#endif
#endif

// ---------------------------------------------------------------------------
// Kernel 1: pack [W_ih ; W_hh] into bf16, n-major [512][192]; bias sum.
// ---------------------------------------------------------------------------
__global__ void lstm_prep_kernel(const float* __restrict__ W_ih,
                                 const float* __restrict__ W_hh,
                                 const float* __restrict__ b_ih,
                                 const float* __restrict__ b_hh,
                                 unsigned short* __restrict__ Wc,
                                 float* __restrict__ bs) {
  int gid = blockIdx.x * blockDim.x + threadIdx.x;
  if (gid < G_DIM) bs[gid] = b_ih[gid] + b_hh[gid];
  if (gid < G_DIM * K_DIM) {
    int n = gid / K_DIM;
    int k = gid - n * K_DIM;
    float v = (k < I_DIM) ? W_ih[n * I_DIM + k] : W_hh[n * H_DIM + (k - I_DIM)];
    unsigned int u = __builtin_bit_cast(unsigned int, v);
    unsigned int r = (u + 0x7FFFu + ((u >> 16) & 1u)) >> 16;  // RNE to bf16
    Wc[(size_t)n * K_DIM + k] = (unsigned short)r;
  }
}

// K index held by packed-pair j for this half-lane (ISA 16-bit A/B layout)
__device__ __forceinline__ int kpair(int j, int hi) {
  return ((j < 4) ? (2 * j) : (16 + 2 * (j - 4))) + (hi ? 8 : 0);
}

// Issue async stage of one 16x64 f32 x-tile into LDS (threads 0..255,
// 16 bytes each; thread tid owns row tid>>4, cols (tid&15)*4 .. +3).
__device__ __forceinline__ void fetch_x(const float* src, float* dstLds, int tid) {
  if (tid < 256) {
    int r  = tid >> 4;
    int cg = (tid & 15) * 4;
    const float* g = src + (size_t)r * (T_TOT * I_DIM) + cg;
    float* l = dstLds + r * I_DIM + cg;
#ifdef USE_ASYNC_LDS
    __builtin_amdgcn_global_load_async_to_lds_b128(
        (v4i_as1*)(uintptr_t)g,
        (v4i_as3*)(unsigned int)(uintptr_t)l,
        0, 0);
#else
    float4 v = *(const float4*)g;
    *(float4*)l = v;
#endif
  }
}

// Wait for own async transfers, then convert the 4 floats this thread staged
// into bf16 inside xh.  Self-consistent per thread: no barrier needed between
// the wait and the reads.
__device__ __forceinline__ void convert_own_x(const float* stage,
                                              __bf16 (*xh)[XH_STRIDE], int tid) {
#ifdef USE_ASYNC_LDS
  __builtin_amdgcn_s_wait_asynccnt(0);
#endif
  if (tid < 256) {
    int r = tid >> 4;
    int c = (tid & 15) * 4;
    float4 v = *(const float4*)(stage + r * I_DIM + c);
    v4bf b = {(__bf16)v.x, (__bf16)v.y, (__bf16)v.z, (__bf16)v.w};
    *(v4bf*)&xh[r][c] = b;   // 8-byte aligned (c multiple of 4)
  }
}

// ---------------------------------------------------------------------------
// Kernel 2: persistent fused LSTM. 32 blocks x 512 threads (16 waves).
// Each block owns 16 batch rows for all 1024 timesteps.
// ---------------------------------------------------------------------------
__global__ __launch_bounds__(512)
void lstm_kernel(const float* __restrict__ x,
                 const unsigned short* __restrict__ Wc,
                 const float* __restrict__ bs,
                 float* __restrict__ hlast) {
  __shared__ __align__(16) __bf16 xh[16][XH_STRIDE];      // [x_t | h] bf16
  __shared__ __align__(16) float  xstage[2][16][I_DIM];   // double-buffered x
  __shared__ __align__(16) float  gates[16][G_DIM];       // pre-activation
  __shared__ float bsl[G_DIM];

  const int tid  = threadIdx.x;
  const int lane = tid & 31;
  const int wave = tid >> 5;
  const int m    = lane & 15;
  const int hi   = lane >> 4;
  const int b0   = blockIdx.x * 16;

  if (tid < G_DIM) bsl[tid] = bs[tid];

  // Per-thread cell state: row mu, 4 consecutive hidden units at u0.
  const int mu = tid >> 5;
  const int u0 = (tid & 31) * 4;
  float creg[4] = {0.f, 0.f, 0.f, 0.f};
#pragma unroll
  for (int q = 0; q < 4; ++q) xh[mu][I_DIM + u0 + q] = (__bf16)0.f;

  // Load this wave's 12 resident B-fragments (2 N-tiles x 6 K-slices).
  v16bf bfr[2][6];
#pragma unroll
  for (int tIdx = 0; tIdx < 2; ++tIdx) {
    int n = (wave * 2 + tIdx) * 16 + m;   // global gate column for this lane
#pragma unroll
    for (int s = 0; s < 6; ++s) {
      v8u p;
#pragma unroll
      for (int j = 0; j < 8; ++j) {
        int k = 32 * s + kpair(j, hi);
        p[j] = *(const unsigned int*)(Wc + (size_t)n * K_DIM + k);
      }
      bfr[tIdx][s] = __builtin_bit_cast(v16bf, p);
    }
  }

  // Preamble: stage+convert x[0], prefetch x[1], publish xh.
  fetch_x(x + (size_t)b0 * (T_TOT * I_DIM), &xstage[0][0][0], tid);
  convert_own_x(&xstage[0][0][0], xh, tid);
  fetch_x(x + ((size_t)b0 * T_TOT + 1) * I_DIM, &xstage[1][0][0], tid);
  __syncthreads();  // xh = [x_0 | h_0] visible to all

  for (int t = 0; t < T_TOT; ++t) {
    // ---- gate GEMM: this wave's 32 columns of [x_t|h] x Wc^T --------------
    // Load ALL A-fragments into distinct registers first so the 12
    // ds_load_b128 can pipeline instead of serializing behind each WMMA pair.
    v16bf av[6];
#pragma unroll
    for (int s = 0; s < 6; ++s) {
      v8u a;
#pragma unroll
      for (int j = 0; j < 8; ++j) {
        int k = 32 * s + kpair(j, hi);
        a[j] = *(const unsigned int*)&xh[m][k];
      }
      av[s] = __builtin_bit_cast(v16bf, a);
    }
    // Split even/odd K-slices into separate accumulators to halve the WMMA
    // dependency-chain depth.
    v8f acc0a = {}, acc0b = {}, acc1a = {}, acc1b = {};
#pragma unroll
    for (int s = 0; s < 6; ++s) {
      if (s & 1) {
        acc0b = __builtin_amdgcn_wmma_f32_16x16x32_bf16(false, av[s], false, bfr[0][s],
                                                        (short)0, acc0b, false, false);
        acc1b = __builtin_amdgcn_wmma_f32_16x16x32_bf16(false, av[s], false, bfr[1][s],
                                                        (short)0, acc1b, false, false);
      } else {
        acc0a = __builtin_amdgcn_wmma_f32_16x16x32_bf16(false, av[s], false, bfr[0][s],
                                                        (short)0, acc0a, false, false);
        acc1a = __builtin_amdgcn_wmma_f32_16x16x32_bf16(false, av[s], false, bfr[1][s],
                                                        (short)0, acc1a, false, false);
      }
    }
    {
      v8f acc0 = acc0a + acc0b;
      v8f acc1 = acc1a + acc1b;
      int col0 = (wave * 2 + 0) * 16 + m;
      int col1 = (wave * 2 + 1) * 16 + m;
#pragma unroll
      for (int r = 0; r < 8; ++r) {     // C/D layout: VGPR r -> M = r + 8*hi
        int mr = r + hi * 8;
        gates[mr][col0] = acc0[r];
        gates[mr][col1] = acc1[r];
      }
    }
    __syncthreads();  // [A] gates ready; all xh reads for step t done

    // ---- elementwise LSTM cell update for 4 owned units -------------------
    float hv[4];
#pragma unroll
    for (int q = 0; q < 4; ++q) {
      int u = u0 + q;
      float ig = gates[mu][u]       + bsl[u];
      float fg = gates[mu][u + 128] + bsl[u + 128];
      float gg = gates[mu][u + 256] + bsl[u + 256];
      float og = gates[mu][u + 384] + bsl[u + 384];
      ig = 1.f / (1.f + __expf(-ig));
      fg = 1.f / (1.f + __expf(-fg));
      og = 1.f / (1.f + __expf(-og));
      float e2 = __expf(2.f * gg);
      gg = (e2 - 1.f) / (e2 + 1.f);
      float c = fg * creg[q] + ig * gg;
      creg[q] = c;
      float ec = __expf(2.f * c);
      hv[q] = og * ((ec - 1.f) / (ec + 1.f));
      xh[mu][I_DIM + u] = (__bf16)hv[q];
    }
    if (t == T_TOT - 1) {
#pragma unroll
      for (int q = 0; q < 4; ++q)
        hlast[(size_t)(b0 + mu) * H_DIM + u0 + q] = hv[q];
    } else {
      // Convert x[t+1] (staged during step t-1) and prefetch x[t+2].
      convert_own_x(&xstage[(t + 1) & 1][0][0], xh, tid);
      if (t + 2 < T_TOT)
        fetch_x(x + ((size_t)b0 * T_TOT + (t + 2)) * I_DIM,
                &xstage[t & 1][0][0], tid);
    }
    __syncthreads();  // [B] xh = [x_{t+1} | h_{t+1}] visible to all
  }
}

// ---------------------------------------------------------------------------
// Kernel 3: y = (h_last @ W1^T + b1) @ W2^T + b2. One block per batch row.
// ---------------------------------------------------------------------------
__global__ __launch_bounds__(256)
void fc_kernel(const float* __restrict__ hlast,
               const float* __restrict__ W1, const float* __restrict__ b1,
               const float* __restrict__ W2, const float* __restrict__ b2,
               float* __restrict__ out) {
  __shared__ float red[256];
  int b = blockIdx.x, j = threadIdx.x;
  const float* h = hlast + (size_t)b * H_DIM;
  const float* w = W1 + (size_t)j * H_DIM;
  float s = b1[j];
#pragma unroll 4
  for (int k = 0; k < H_DIM; ++k) s += h[k] * w[k];
  red[j] = s * W2[j];
  __syncthreads();
  for (int off = 128; off > 0; off >>= 1) {
    if (j < off) red[j] += red[j + off];
    __syncthreads();
  }
  if (j == 0) out[b] = red[0] + b2[0];
}

// ---------------------------------------------------------------------------
extern "C" void kernel_launch(void* const* d_in, const int* in_sizes, int n_in,
                              void* d_out, int out_size, void* d_ws, size_t ws_size,
                              hipStream_t stream) {
  (void)in_sizes; (void)n_in; (void)out_size; (void)ws_size;
  const float* x    = (const float*)d_in[0];
  const float* W_ih = (const float*)d_in[1];
  const float* W_hh = (const float*)d_in[2];
  const float* b_ih = (const float*)d_in[3];
  const float* b_hh = (const float*)d_in[4];
  const float* W1   = (const float*)d_in[5];
  const float* b1   = (const float*)d_in[6];
  const float* W2   = (const float*)d_in[7];
  const float* b2   = (const float*)d_in[8];
  float* out = (float*)d_out;

  // Workspace layout (all 256B aligned):
  //   [0, 196608)        Wc   bf16 [512][192]
  //   [196608, 198656)   bs   f32  [512]
  //   [198912, 461056)   hlast f32 [512][128]
  char* ws = (char*)d_ws;
  unsigned short* Wc = (unsigned short*)(ws);
  float* bs    = (float*)(ws + 196608);
  float* hlast = (float*)(ws + 198912);

  lstm_prep_kernel<<<(G_DIM * K_DIM + 255) / 256, 256, 0, stream>>>(
      W_ih, W_hh, b_ih, b_hh, Wc, bs);
  lstm_kernel<<<B_TOT / 16, 512, 0, stream>>>(x, Wc, bs, hlast);
  fc_kernel<<<B_TOT, 256, 0, stream>>>(hlast, W1, b1, W2, b2, out);
}
